// AxialAttentionBlock_80573586473341
// MI455X (gfx1250) — matrix operand order repaired and checked by reading, not compiled, not run on hardware
//
#include <hip/hip_runtime.h>

// ---------------- problem constants ----------------
#define HIDC   256
#define NHC    8
#define HDC    32
#define NTOK   131072          // 2*256*256 tokens
#define NFUSED 1792            // 7*HID
#define FFC    1024            // 4*HID

typedef __attribute__((ext_vector_type(16))) __bf16 v16bf;
typedef __attribute__((ext_vector_type(8)))  __bf16 v8bf;
typedef __attribute__((ext_vector_type(8)))  float  v8f;

__device__ __forceinline__ v8f wmma_bf16(v16bf a, v16bf b, v8f c) {
  // D = A(16x32 bf16) * B(32x16 bf16) + C(16x16 f32)
  return __builtin_amdgcn_wmma_f32_16x16x32_bf16(false, a, false, b, (short)0, c,
                                                 false, false);
}

__device__ __forceinline__ v16bf zero16() {
  v16bf r;
#pragma unroll
  for (int i = 0; i < 16; ++i) r[i] = (__bf16)0.0f;
  return r;
}
__device__ __forceinline__ v8f zero8f() {
  v8f r;
#pragma unroll
  for (int i = 0; i < 8; ++i) r[i] = 0.0f;
  return r;
}
// Concatenate two contiguous 8-element chunks into a 16-element fragment.
__device__ __forceinline__ v16bf ld_frag(const __bf16* p0, const __bf16* p1) {
  v8bf lo = *(const v8bf*)p0;
  v8bf hi = *(const v8bf*)p1;
  return __builtin_shufflevector(lo, hi, 0, 1, 2, 3, 4, 5, 6, 7, 8, 9, 10, 11, 12,
                                 13, 14, 15);
}

__device__ __forceinline__ float gelu_tanh(float x) {
  float u = 0.7978845608028654f * (x + 0.044715f * x * x * x);
  return 0.5f * x * (1.0f + tanhf(u));
}

// gfx1250 async global->LDS DMA (tracked by ASYNCcnt), 16B per lane.
__device__ __forceinline__ void async_copy_b128(unsigned lds_byte_addr,
                                                const void* gaddr) {
  unsigned long long ga = (unsigned long long)(uintptr_t)gaddr;
  asm volatile("global_load_async_to_lds_b128 %0, %1, off"
               :
               : "v"(lds_byte_addr), "v"(ga)
               : "memory");
}
__device__ __forceinline__ void wait_asynccnt0() {
  asm volatile("s_wait_asynccnt 0x0" ::: "memory");
}

// ---------------- small utility kernels ----------------
__global__ void cvt_f32_bf16_kernel(const float* __restrict__ src,
                                    __bf16* __restrict__ dst, long n) {
  long i = (long)blockIdx.x * 256 + threadIdx.x;
  if (i < n) dst[i] = (__bf16)src[i];
}

// LayerNorm over HID=256 channels, scale only; one wave per token, cast to bf16.
__global__ __launch_bounds__(256) void ln0_kernel(const float* __restrict__ x,
                                                  const float* __restrict__ scale,
                                                  __bf16* __restrict__ h) {
  int wv = threadIdx.x >> 5, lane = threadIdx.x & 31;
  long tok = (long)blockIdx.x * 8 + wv;
  const float* xp = x + tok * HIDC;
  float v[8];
  float s = 0.0f;
#pragma unroll
  for (int j = 0; j < 8; ++j) { v[j] = xp[lane + 32 * j]; s += v[j]; }
#pragma unroll
  for (int m = 16; m >= 1; m >>= 1) s += __shfl_xor(s, m, 32);
  float mu = s * (1.0f / 256.0f);
  float vs = 0.0f;
#pragma unroll
  for (int j = 0; j < 8; ++j) { float d = v[j] - mu; vs += d * d; }
#pragma unroll
  for (int m = 16; m >= 1; m >>= 1) vs += __shfl_xor(vs, m, 32);
  float rstd = rsqrtf(vs * (1.0f / 256.0f) + 1e-6f);
  __bf16* hp = h + tok * HIDC;
#pragma unroll
  for (int j = 0; j < 8; ++j) {
    int c = lane + 32 * j;
    hp[c] = (__bf16)((v[j] - mu) * rstd * scale[c]);
  }
}

// Per-(token,head) LayerNorm over HD=32 (in-place on bf16 buffer). Wave = head.
__global__ __launch_bounds__(256) void lnhd_kernel(__bf16* __restrict__ buf,
                                                   const float* __restrict__ scale) {
  int wv = threadIdx.x >> 5, d = threadIdx.x & 31;
  long tok = blockIdx.x;
  __bf16* p = buf + tok * HIDC + wv * HDC;
  float v = (float)p[d];
  float s = v;
#pragma unroll
  for (int m = 16; m >= 1; m >>= 1) s += __shfl_xor(s, m, 32);
  float mu = s * (1.0f / 32.0f);
  float dd = v - mu;
  float vs = dd * dd;
#pragma unroll
  for (int m = 16; m >= 1; m >>= 1) vs += __shfl_xor(vs, m, 32);
  float rstd = rsqrtf(vs * (1.0f / 32.0f) + 1e-6f);
  p[d] = (__bf16)(dd * rstd * scale[d]);
}

// ---------------- tiled bf16 WMMA GEMM ----------------
// C(MxN) = A(MxK) * B(KxN); block tile 128x128x32, 8 waves (2x4 tiles of 16x16/wave)
// MODE 0: C=acc   MODE 1: C+=acc   MODE 2: fused-proj routing (q/k/v/gelu(ff))
#define BM 128
#define BN 128
#define BK 32
#define SAS 48  // LDS row stride (elements): 96B, multiple of 16B, staggers banks

template <int MODE>
__global__ __launch_bounds__(256) void gemm_bf16_kernel(
    const __bf16* __restrict__ A, long lda, const __bf16* __restrict__ Bm, long ldb,
    long M, long N, long K, float* __restrict__ C, const float* __restrict__ bias,
    __bf16* __restrict__ q_out, __bf16* __restrict__ k_out,
    __bf16* __restrict__ v_out, __bf16* __restrict__ ff_out) {
  __shared__ __bf16 sA[BM * SAS];
  __shared__ __bf16 sBT[BN * SAS];  // stored transposed: [n][k]
  int t = threadIdx.x;
  long m0 = (long)blockIdx.y * BM;
  long n0 = (long)blockIdx.x * BN;
  int wv = t >> 5, lane = t & 31;
  int wrow = wv & 3, wcol = wv >> 2;  // 4x2 wave grid -> 32x64 per wave
  int ml = lane & 15, half = lane >> 4;
  unsigned sA_base = (unsigned)(uintptr_t)&sA[0];

  v8f acc[2][4];
#pragma unroll
  for (int rt = 0; rt < 2; ++rt)
#pragma unroll
    for (int ct = 0; ct < 4; ++ct) acc[rt][ct] = zero8f();

  for (long k0 = 0; k0 < K; k0 += BK) {
    // stage A tile (128x32 bf16) with async global->LDS DMA (no VGPR bounce)
#pragma unroll
    for (int i = 0; i < 2; ++i) {
      int chunk = t + 256 * i;
      int row = chunk >> 2;
      int cc = chunk & 3;
      const __bf16* src = A + (m0 + row) * lda + k0 + cc * 8;
      if (i == 0) __builtin_prefetch(src + BK, 0, 1);
      async_copy_b128(sA_base + (unsigned)(row * SAS + cc * 8) * 2u, src);
    }
    // stage B tile (32x128), transposing into sBT[n][k]
#pragma unroll
    for (int i = 0; i < 2; ++i) {
      int chunk = t + 256 * i;
      int kk = chunk >> 4;
      int nc = chunk & 15;
      const __bf16* src = Bm + (k0 + kk) * ldb + n0 + nc * 8;
      alignas(16) __bf16 e[8];
      *(uint4*)e = *(const uint4*)src;
#pragma unroll
      for (int j = 0; j < 8; ++j) sBT[(nc * 8 + j) * SAS + kk] = e[j];
    }
    wait_asynccnt0();  // this wave's async LDS writes retired
    __syncthreads();   // compiler adds s_wait_dscnt for the B ds_stores + barrier

    v16bf afr[2];
#pragma unroll
    for (int rt = 0; rt < 2; ++rt) {
      // A layout: lane m<16 -> K{0..7,16..23}; lane m+16 -> K{8..15,24..31}
      const __bf16* p = &sA[(wrow * 32 + rt * 16 + ml) * SAS + half * 8];
      afr[rt] = ld_frag(p, p + 16);
    }
    v16bf bfr[4];
#pragma unroll
    for (int ct = 0; ct < 4; ++ct) {
      // B layout: lane n<16 -> K{0..7,16..23} ; lane n+16 -> K{8..15,24..31}
      const __bf16* p = &sBT[(wcol * 64 + ct * 16 + ml) * SAS + half * 8];
      bfr[ct] = ld_frag(p, p + 16);
    }
#pragma unroll
    for (int rt = 0; rt < 2; ++rt)
#pragma unroll
      for (int ct = 0; ct < 4; ++ct)
        acc[rt][ct] = wmma_bf16(afr[rt], bfr[ct], acc[rt][ct]);
    __syncthreads();
  }

  // epilogue (C layout: slot r, lanes0-15 -> M=r, lanes16-31 -> M=8+r, N=ml)
#pragma unroll
  for (int rt = 0; rt < 2; ++rt) {
    long rbase = m0 + wrow * 32 + rt * 16 + half * 8;
#pragma unroll
    for (int ct = 0; ct < 4; ++ct) {
      long n = n0 + wcol * 64 + ct * 16 + ml;
#pragma unroll
      for (int r = 0; r < 8; ++r) {
        long row = rbase + r;
        float val = acc[rt][ct][r];
        if (MODE == 0) {
          C[row * N + n] = val;
        } else if (MODE == 1) {
          C[row * N + n] += val;
        } else {
          val += bias[n];
          if (n < 256)
            q_out[row * 256 + n] = (__bf16)val;
          else if (n < 512)
            k_out[row * 256 + (n - 256)] = (__bf16)val;
          else if (n < 768)
            v_out[row * 256 + (n - 512)] = (__bf16)val;
          else
            ff_out[row * 1024 + (n - 768)] = (__bf16)gelu_tanh(val);
        }
      }
    }
  }
}

// ---------------- per-token head attention (faithful to reference einsum) ------
// The reference's transpose-then-'bqhd,bkhd->bhqk' einsum attends over HEADS at
// each spatial position; attn_h == attn_w, so: o = 2*softmax(q k^T/sqrt(32)) v
// with q,k,v in R^{8x32} per token. One wave per token; heads packed into a
// zero-padded 16x16x32 WMMA.
__global__ __launch_bounds__(256) void headattn_kernel(const __bf16* __restrict__ q,
                                                       const __bf16* __restrict__ k,
                                                       const __bf16* __restrict__ v,
                                                       __bf16* __restrict__ out) {
  __shared__ __bf16 sP[8][16][32];  // per-wave P in A-matrix layout source form
  int t = threadIdx.x, wv = t >> 5, lane = t & 31;
  int ml = lane & 15, half = lane >> 4;
  // zero the padded P regions once
  {
    __bf16* pz = &sP[0][0][0];
    for (int i = t; i < 8 * 16 * 32; i += 256) pz[i] = (__bf16)0.0f;
  }
  __syncthreads();

  long tok = (long)blockIdx.x * 8 + wv;
  const __bf16* qp = q + tok * HIDC;
  const __bf16* kp = k + tok * HIDC;
  const __bf16* vp = v + tok * HIDC;

  // A = Q (rows = heads 0..7, K = d 0..31), zero-padded rows 8..15
  v16bf aq = zero16();
  if (ml < 8) aq = ld_frag(qp + ml * 32 + half * 8, qp + ml * 32 + 16 + half * 8);
  // B = K^T (lane n: K[n][half*8+0..7], K[n][16+half*8+0..7]), zero-padded cols 8..15
  v16bf bk = zero16();
  if (ml < 8) bk = ld_frag(kp + ml * 32 + half * 8, kp + ml * 32 + 16 + half * 8);

  v8f s = wmma_bf16(aq, bk, zero8f());

  const float scl = 0.17677669529663687f;  // 1/sqrt(32)
  float pbuf[8], rinv[8];
#pragma unroll
  for (int r = 0; r < 8; ++r) {
    float val = (ml < 8) ? s[r] * scl : -3.0e38f;
    float mx = val;
#pragma unroll
    for (int m = 1; m <= 8; m <<= 1) mx = fmaxf(mx, __shfl_xor(mx, m, 16));
    float p = (ml < 8) ? __expf(val - mx) : 0.0f;
    float sm = p;
#pragma unroll
    for (int m = 1; m <= 8; m <<= 1) sm += __shfl_xor(sm, m, 16);
    pbuf[r] = p;
    rinv[r] = 1.0f / sm;
  }
  // store valid 8x8 P (C-layout -> row-major) into zero-padded LDS tile
  if (lane < 8) {
#pragma unroll
    for (int r = 0; r < 8; ++r) sP[wv][r][lane] = (__bf16)pbuf[r];
  }
  __syncthreads();  // uniform; guarantees ds store->load ordering for the reload

  // A = P (16x32, K = key-head, zero padded); B = V (K=head kk, N=d column)
  const __bf16* sprow = &sP[wv][ml][0];
  v16bf ap = ld_frag(sprow + half * 8, sprow + half * 8 + 16);
  __bf16* outp = out + tok * HIDC;
#pragma unroll
  for (int ct = 0; ct < 2; ++ct) {
    int d0 = ct * 16;
    v16bf bv = zero16();
    if (half == 0) {
#pragma unroll
      for (int j = 0; j < 8; ++j) bv[j] = vp[j * 32 + d0 + ml];
    }
    v8f o = wmma_bf16(ap, bv, zero8f());
    if (half == 0) {
#pragma unroll
      for (int r = 0; r < 8; ++r)
        outp[(long)r * 32 + d0 + ml] = (__bf16)(2.0f * o[r] * rinv[r]);
    }
  }
}

// ---------------- final residual epilogue ----------------
__global__ __launch_bounds__(256) void final_kernel(
    const float* __restrict__ x, const float* __restrict__ oacc,
    const float* __restrict__ b_out, const float* __restrict__ b_mlp,
    const float* __restrict__ gamma, float* __restrict__ out) {
  long i = ((long)blockIdx.x * 256 + threadIdx.x) * 4;
  int c = (int)(i & (HIDC - 1));
  float4 xv = *(const float4*)(x + i);
  float4 ov = *(const float4*)(oacc + i);
  float4 r;
  r.x = xv.x + gamma[c + 0] * (ov.x + b_out[c + 0] + b_mlp[c + 0]);
  r.y = xv.y + gamma[c + 1] * (ov.y + b_out[c + 1] + b_mlp[c + 1]);
  r.z = xv.z + gamma[c + 2] * (ov.z + b_out[c + 2] + b_mlp[c + 2]);
  r.w = xv.w + gamma[c + 3] * (ov.w + b_out[c + 3] + b_mlp[c + 3]);
  *(float4*)(out + i) = r;
}

// ---------------- host launcher ----------------
extern "C" void kernel_launch(void* const* d_in, const int* in_sizes, int n_in,
                              void* d_out, int out_size, void* d_ws, size_t ws_size,
                              hipStream_t stream) {
  (void)in_sizes; (void)n_in; (void)out_size; (void)ws_size;
  const float* x       = (const float*)d_in[0];
  const float* ln0_s   = (const float*)d_in[1];
  const float* w_fused = (const float*)d_in[2];
  const float* b_fused = (const float*)d_in[3];
  const float* lnq_s   = (const float*)d_in[4];
  const float* lnk_s   = (const float*)d_in[5];
  const float* w_out   = (const float*)d_in[6];
  const float* b_out   = (const float*)d_in[7];
  const float* w_mlp   = (const float*)d_in[8];
  const float* b_mlp   = (const float*)d_in[9];
  const float* gamma   = (const float*)d_in[10];
  float* out = (float*)d_out;

  char* ws = (char*)d_ws;
  size_t off = 0;
  auto alloc = [&](size_t bytes) -> char* {
    char* p = ws + off;
    off += (bytes + 255) & ~(size_t)255;
    return p;
  };
  __bf16* h_bf    = (__bf16*)alloc((size_t)NTOK * HIDC * 2);
  __bf16* q_bf    = (__bf16*)alloc((size_t)NTOK * HIDC * 2);
  __bf16* k_bf    = (__bf16*)alloc((size_t)NTOK * HIDC * 2);
  __bf16* v_bf    = (__bf16*)alloc((size_t)NTOK * HIDC * 2);
  __bf16* ff_bf   = (__bf16*)alloc((size_t)NTOK * FFC * 2);
  __bf16* attn_bf = (__bf16*)alloc((size_t)NTOK * HIDC * 2);
  float*  oacc    = (float*)alloc((size_t)NTOK * HIDC * 4);
  __bf16* wf_bf   = (__bf16*)alloc((size_t)HIDC * NFUSED * 2);
  __bf16* wo_bf   = (__bf16*)alloc((size_t)HIDC * HIDC * 2);
  __bf16* wm_bf   = (__bf16*)alloc((size_t)FFC * HIDC * 2);

  // weights -> bf16
  cvt_f32_bf16_kernel<<<(HIDC * NFUSED + 255) / 256, 256, 0, stream>>>(
      w_fused, wf_bf, (long)HIDC * NFUSED);
  cvt_f32_bf16_kernel<<<(HIDC * HIDC + 255) / 256, 256, 0, stream>>>(
      w_out, wo_bf, (long)HIDC * HIDC);
  cvt_f32_bf16_kernel<<<(FFC * HIDC + 255) / 256, 256, 0, stream>>>(
      w_mlp, wm_bf, (long)FFC * HIDC);

  // LN0 + cast
  ln0_kernel<<<NTOK / 8, 256, 0, stream>>>(x, ln0_s, h_bf);

  // fused projection GEMM: (NTOK x 256) @ (256 x 1792), epilogue routes q/k/v/ff
  gemm_bf16_kernel<2><<<dim3(NFUSED / BN, NTOK / BM), 256, 0, stream>>>(
      h_bf, 256, wf_bf, NFUSED, (long)NTOK, (long)NFUSED, 256, nullptr, b_fused,
      q_bf, k_bf, v_bf, ff_bf);

  // per-head LayerNorm on q, k (in place)
  lnhd_kernel<<<NTOK, 256, 0, stream>>>(q_bf, lnq_s);
  lnhd_kernel<<<NTOK, 256, 0, stream>>>(k_bf, lnk_s);

  // per-token head attention (includes the x2 from attn_h + attn_w)
  headattn_kernel<<<NTOK / 8, 256, 0, stream>>>(q_bf, k_bf, v_bf, attn_bf);

  // out projection: oacc = attn @ w_out
  gemm_bf16_kernel<0><<<dim3(HIDC / BN, NTOK / BM), 256, 0, stream>>>(
      attn_bf, 256, wo_bf, HIDC, (long)NTOK, (long)HIDC, 256, oacc, nullptr,
      nullptr, nullptr, nullptr, nullptr);

  // MLP: oacc += gelu(ff) @ w_mlp
  gemm_bf16_kernel<1><<<dim3(HIDC / BN, NTOK / BM), 256, 0, stream>>>(
      ff_bf, FFC, wm_bf, HIDC, (long)NTOK, (long)HIDC, (long)FFC, oacc, nullptr,
      nullptr, nullptr, nullptr, nullptr);

  // final: out = x + gamma * (oacc + b_out + b_mlp)
  final_kernel<<<((long)NTOK * HIDC / 4) / 256, 256, 0, stream>>>(x, oacc, b_out,
                                                                  b_mlp, gamma, out);
}
